// MultiHeadAttentionWithBias_53618371723530
// MI455X (gfx1250) — matrix operand order repaired and checked by reading, not compiled, not run on hardware
//
#include <hip/hip_runtime.h>
#include <hip/hip_bf16.h>

#define S_LEN   2048
#define BATCH   2
#define DM      1024
#define NHEAD   16
#define HDIM    64
#define BS      (BATCH * S_LEN)      /* 4096 rows */
#define NEG_BIG (-1.0e9f)

typedef __attribute__((ext_vector_type(16))) __bf16   bf16x16;
typedef __attribute__((ext_vector_type(8)))  float    f32x8;
typedef __attribute__((ext_vector_type(4)))  unsigned u32x4;
typedef __attribute__((ext_vector_type(4)))  unsigned tdm_g0_t;
typedef __attribute__((ext_vector_type(8)))  int      tdm_g1_t;
typedef __attribute__((ext_vector_type(4)))  int      tdm_g2_t;

union F8       { f32x8 v; float f[8]; };
union BF16Frag { bf16x16 v; u32x4 q[2]; unsigned short h[16]; };

static __device__ __forceinline__ unsigned short f2bf(float x) {
  unsigned u = __float_as_uint(x);
  unsigned r = u + 0x7FFFu + ((u >> 16) & 1u);   // round-to-nearest-even
  return (unsigned short)(r >> 16);
}

static __device__ __forceinline__ f32x8 zero8() {
  F8 z;
#pragma unroll
  for (int i = 0; i < 8; ++i) z.f[i] = 0.0f;
  return z.v;
}

// LDS byte offset of a __shared__-derived generic pointer: flat LDS addresses
// carry the LDS offset in addr[31:0] (aperture in the high half).
static __device__ __forceinline__ unsigned lds_off(const void* p) {
  return (unsigned)(unsigned long long)p;
}

// Issue one 2-D TDM tile load: global (row-major, stride_elems, bf16) -> LDS packed
// [tile_h][tile_w]. Descriptor per CDNA5 ISA 8.3/8.4. Wave-level op (EXEC ignored);
// caller gates to a single wave. Tracked by TENSORcnt.
static __device__ __forceinline__ void tdm_load_2d(unsigned lds_byte_off,
                                                   const unsigned short* gptr,
                                                   unsigned tile_w, unsigned tile_h,
                                                   unsigned stride_elems) {
  unsigned long long ga = (unsigned long long)gptr;
  tdm_g0_t g0;
  g0.x = 1u;                                             // count=1, user descriptor
  g0.y = lds_byte_off;                                   // lds_addr (bytes)
  g0.z = (unsigned)(ga & 0xffffffffu);                   // global_addr[31:0]
  g0.w = (unsigned)((ga >> 32) & 0x01ffffffu) | 0x80000000u; // addr[56:32] | type=2
  const unsigned d0 = tile_w;                            // tensor_dim0 == tile_dim0
  const unsigned d1 = tile_h;                            // tensor_dim1 == tile_dim1
  tdm_g1_t g1;
  g1[0] = (int)(1u << 16);                               // data_size=1 (2 bytes)
  g1[1] = (int)((d0 & 0xffffu) << 16);                   // tensor_dim0[15:0]
  g1[2] = (int)((d0 >> 16) | ((d1 & 0xffffu) << 16));    // dim0 hi | dim1 lo
  g1[3] = (int)((d1 >> 16) | (tile_w << 16));            // dim1 hi | tile_dim0
  g1[4] = (int)(tile_h & 0xffffu);                       // tile_dim1 (tile_dim2=0)
  g1[5] = (int)stride_elems;                             // tensor_dim0_stride[31:0]
  g1[6] = 0;
  g1[7] = 0;
  tdm_g2_t z4 = {0, 0, 0, 0};
#if defined(__clang_major__) && (__clang_major__ >= 23)
  tdm_g1_t z8 = {0, 0, 0, 0, 0, 0, 0, 0};
  __builtin_amdgcn_tensor_load_to_lds(g0, g1, z4, z4, z8, 0);
#else
  __builtin_amdgcn_tensor_load_to_lds(g0, g1, z4, z4, 0);
#endif
}

// A-operand 16x32 bf16 from row-major src (rows=M, cols=K), ld in elements.
static __device__ __forceinline__ bf16x16 load_frag_a(const unsigned short* base,
                                                      int row0, int k0, int ld) {
  const int lane = threadIdx.x & 31;
  const int hf   = lane >> 4;
  const int m    = lane & 15;
  const unsigned short* p = base + (size_t)(row0 + m) * ld + k0 + hf * 8;
  BF16Frag u;
  u.q[0] = *(const u32x4*)(p);
  u.q[1] = *(const u32x4*)(p + 16);
  return u.v;
}

// B-operand 32x16 bf16 from n-major src (rows=N, cols=K), ld in elements.
static __device__ __forceinline__ bf16x16 load_frag_b(const unsigned short* base,
                                                      int n0, int k0, int ld) {
  const int lane = threadIdx.x & 31;
  const int hf   = lane >> 4;
  const int n    = lane & 15;
  const unsigned short* p = base + (size_t)(n0 + n) * ld + k0 + hf * 8;
  BF16Frag u;
  u.q[0] = *(const u32x4*)(p);
  u.q[1] = *(const u32x4*)(p + 16);
  return u.v;
}

static __device__ __forceinline__ f32x8 wmma_bf16(bf16x16 a, bf16x16 b, f32x8 c) {
  return __builtin_amdgcn_wmma_f32_16x16x32_bf16(false, a, false, b, (short)0, c, false, false);
}

static __device__ __forceinline__ float hmax16(float v) {
#pragma unroll
  for (int off = 1; off < 16; off <<= 1) v = fmaxf(v, __shfl_xor(v, off, 32));
  return v;
}
static __device__ __forceinline__ float hsum16(float v) {
#pragma unroll
  for (int off = 1; off < 16; off <<= 1) v += __shfl_xor(v, off, 32);
  return v;
}

// ---------------- conversion kernels ----------------
__global__ void cvt_x_kernel(const float* __restrict__ x, unsigned short* __restrict__ xb) {
  size_t i = (size_t)blockIdx.x * 256 + threadIdx.x;
  xb[i] = f2bf(x[i]);
}

// Transpose + pack weights: wqkvt is [3*DM rows (n)][DM cols (k)] n-major; wot likewise.
__global__ void cvt_w_kernel(const float* __restrict__ wq, const float* __restrict__ wk,
                             const float* __restrict__ wv, const float* __restrict__ wo,
                             unsigned short* __restrict__ wqkvt, unsigned short* __restrict__ wot) {
  size_t i = (size_t)blockIdx.x * 256 + threadIdx.x;     // over DM*DM, i = k*DM + n
  int k = (int)(i >> 10);
  int n = (int)(i & 1023);
  size_t t = (size_t)n * DM + k;
  wqkvt[t]                       = f2bf(wq[i]);
  wqkvt[(size_t)DM * DM + t]     = f2bf(wk[i]);
  wqkvt[(size_t)2 * DM * DM + t] = f2bf(wv[i]);
  wot[t]                         = f2bf(wo[i]);
}

// ---------------- QKV projection GEMM: [4096x1024] x [1024x3072] ----------------
// Block = 128 thr (4 waves). Tile 128(M) x 64(N), K-step 32, TDM double-buffered LDS.
__global__ void gemm_qkv_kernel(const unsigned short* __restrict__ xb,
                                const unsigned short* __restrict__ wt,
                                const float* __restrict__ bqp, const float* __restrict__ bkp,
                                const float* __restrict__ bvp,
                                unsigned short* __restrict__ qout,
                                unsigned short* __restrict__ kout,
                                unsigned short* __restrict__ vtout) {
  __shared__ __align__(16) unsigned short Atile[2][128][32];
  __shared__ __align__(16) unsigned short Btile[2][64][32];
  const int wave = threadIdx.x >> 5;
  const int lane = threadIdx.x & 31;
  const int hf   = lane >> 4;
  const int n    = lane & 15;
  const int m0   = blockIdx.x * 128;
  const int n0   = blockIdx.y * 64;

  F8 acc[8];
#pragma unroll
  for (int j = 0; j < 8; ++j) acc[j].v = zero8();

  if (threadIdx.x < 32) {
    tdm_load_2d(lds_off(&Atile[0][0][0]), xb + (size_t)m0 * DM, 32, 128, DM);
    tdm_load_2d(lds_off(&Btile[0][0][0]), wt + (size_t)n0 * DM, 32, 64, DM);
  }
  int cur = 0;
  for (int k0 = 0; k0 < DM; k0 += 32) {
    if (threadIdx.x < 32) {
      if (k0 + 32 < DM) {
        tdm_load_2d(lds_off(&Atile[cur ^ 1][0][0]), xb + (size_t)m0 * DM + (k0 + 32), 32, 128, DM);
        tdm_load_2d(lds_off(&Btile[cur ^ 1][0][0]), wt + (size_t)n0 * DM + (k0 + 32), 32, 64, DM);
        __builtin_amdgcn_s_wait_tensorcnt(2);
      } else {
        __builtin_amdgcn_s_wait_tensorcnt(0);
      }
    }
    __syncthreads();
    const bf16x16 a0 = load_frag_a(&Atile[cur][0][0], wave * 32, 0, 32);
    const bf16x16 a1 = load_frag_a(&Atile[cur][0][0], wave * 32 + 16, 0, 32);
#pragma unroll
    for (int nt = 0; nt < 4; ++nt) {
      const bf16x16 bfr = load_frag_b(&Btile[cur][0][0], nt * 16, 0, 32);
      acc[nt].v     = wmma_bf16(a0, bfr, acc[nt].v);
      acc[4 + nt].v = wmma_bf16(a1, bfr, acc[4 + nt].v);
    }
    __syncthreads();
    cur ^= 1;
  }

  const int region = n0 >> 10;  // 0=Q 1=K 2=V, uniform per block (n0 multiple of 64)
#pragma unroll
  for (int sub = 0; sub < 2; ++sub) {
#pragma unroll
    for (int nt = 0; nt < 4; ++nt) {
      const int ncol = n0 + nt * 16 + n;
      const int nn = ncol & 1023;
      const int hh = nn >> 6, hd = nn & 63;
      const float bias = (region == 0) ? bqp[nn] : (region == 1) ? bkp[nn] : bvp[nn];
#pragma unroll
      for (int r = 0; r < 8; ++r) {
        const int m = m0 + wave * 32 + sub * 16 + r + hf * 8;
        const int bb = m >> 11;
        const int s  = m & 2047;
        const unsigned short val = f2bf(acc[sub * 4 + nt].f[r] + bias);
        if (region == 0)
          qout[(((size_t)(bb * NHEAD + hh)) * S_LEN + s) * HDIM + hd] = val;
        else if (region == 1)
          kout[(((size_t)(bb * NHEAD + hh)) * S_LEN + s) * HDIM + hd] = val;
        else
          vtout[(((size_t)(bb * NHEAD + hh)) * HDIM + hd) * S_LEN + s] = val;  // V^T
      }
    }
  }
}

// ---------------- flash attention: grid (S/64, H, B), block 128 ----------------
__global__ void attn_kernel(const unsigned short* __restrict__ qg,
                            const unsigned short* __restrict__ kg,
                            const unsigned short* __restrict__ vtg,
                            const int* __restrict__ mask,
                            const float* __restrict__ attn_bias,
                            unsigned short* __restrict__ outg) {
  __shared__ __align__(16) unsigned short Kt[2][32][64];   // [t][hd] n-major for QK^T
  __shared__ __align__(16) unsigned short Vt[2][64][32];   // [hd][t] n-major for PV
  __shared__ __align__(16) unsigned short Plds[4][16][32];
  const int wave = threadIdx.x >> 5;
  const int lane = threadIdx.x & 31;
  const int hf   = lane >> 4;
  const int n    = lane & 15;
  const int bb   = blockIdx.z;
  const int h    = blockIdx.y;
  const int row0 = blockIdx.x * 64 + wave * 16;

  const unsigned short* qb = qg  + ((size_t)(bb * NHEAD + h) * S_LEN) * HDIM;
  const unsigned short* kb = kg  + ((size_t)(bb * NHEAD + h) * S_LEN) * HDIM;
  const unsigned short* vb = vtg + ((size_t)(bb * NHEAD + h) * HDIM) * S_LEN;
  const int* mrow = mask + bb * S_LEN;
  const float hbias = attn_bias[h];

  const bf16x16 qa0 = load_frag_a(qb, row0, 0, HDIM);
  const bf16x16 qa1 = load_frag_a(qb, row0, 32, HDIM);

  F8 acc[4];
#pragma unroll
  for (int j = 0; j < 4; ++j) acc[j].v = zero8();
  float mi[8], li[8];
#pragma unroll
  for (int r = 0; r < 8; ++r) { mi[r] = -3.0e38f; li[r] = 0.0f; }

  if (threadIdx.x < 32) {
    tdm_load_2d(lds_off(&Kt[0][0][0]), kb, 64, 32, HDIM);      // rows t, stride 64
    tdm_load_2d(lds_off(&Vt[0][0][0]), vb, 32, 64, S_LEN);     // rows hd, stride S
  }
  int cur = 0;
  for (int t0 = 0; t0 < S_LEN; t0 += 32) {
    if (threadIdx.x < 32) {
      if (t0 + 32 < S_LEN) {
        tdm_load_2d(lds_off(&Kt[cur ^ 1][0][0]), kb + (size_t)(t0 + 32) * HDIM, 64, 32, HDIM);
        tdm_load_2d(lds_off(&Vt[cur ^ 1][0][0]), vb + (t0 + 32), 32, 64, S_LEN);
        __builtin_amdgcn_s_wait_tensorcnt(2);
      } else {
        __builtin_amdgcn_s_wait_tensorcnt(0);
      }
    }
    __syncthreads();

    F8 sc[2];
#pragma unroll
    for (int nt = 0; nt < 2; ++nt) {
      const bf16x16 kf0 = load_frag_b(&Kt[cur][0][0], nt * 16, 0, 64);
      const bf16x16 kf1 = load_frag_b(&Kt[cur][0][0], nt * 16, 32, 64);
      f32x8 c  = wmma_bf16(qa0, kf0, zero8());
      sc[nt].v = wmma_bf16(qa1, kf1, c);
    }
    const bool keep0 = (mrow[t0 + n] != 0);
    const bool keep1 = (mrow[t0 + 16 + n] != 0);
#pragma unroll
    for (int r = 0; r < 8; ++r) {
      sc[0].f[r] = keep0 ? (sc[0].f[r] * 0.125f + hbias) : NEG_BIG;
      sc[1].f[r] = keep1 ? (sc[1].f[r] * 0.125f + hbias) : NEG_BIG;
    }
#pragma unroll
    for (int r = 0; r < 8; ++r) {
      float v = hmax16(fmaxf(sc[0].f[r], sc[1].f[r]));
      const float mn = fmaxf(mi[r], v);
      const float sf = __expf(mi[r] - mn);
      mi[r] = mn;
      const float p0 = __expf(sc[0].f[r] - mn);
      const float p1 = __expf(sc[1].f[r] - mn);
      li[r] = li[r] * sf + hsum16(p0 + p1);
      Plds[wave][r + hf * 8][n]      = f2bf(p0);
      Plds[wave][r + hf * 8][16 + n] = f2bf(p1);
#pragma unroll
      for (int j = 0; j < 4; ++j) acc[j].f[r] *= sf;
    }
    __syncthreads();
    const bf16x16 pa = load_frag_a(&Plds[wave][0][0], 0, 0, 32);
#pragma unroll
    for (int j = 0; j < 4; ++j) {
      const bf16x16 vfr = load_frag_b(&Vt[cur][0][0], j * 16, 0, 32);
      acc[j].v = wmma_bf16(pa, vfr, acc[j].v);
    }
    __syncthreads();
    cur ^= 1;
  }

#pragma unroll
  for (int r = 0; r < 8; ++r) {
    const int s = row0 + r + hf * 8;
    const float inv = 1.0f / fmaxf(li[r], 1e-20f);
#pragma unroll
    for (int j = 0; j < 4; ++j)
      outg[((size_t)(bb * S_LEN + s)) * DM + h * HDIM + j * 16 + n] = f2bf(acc[j].f[r] * inv);
  }
}

// ---------------- output projection GEMM: [4096x1024] x [1024x1024] -> f32 ----------------
__global__ void gemm_out_kernel(const unsigned short* __restrict__ ao,
                                const unsigned short* __restrict__ wot,
                                const float* __restrict__ bo,
                                float* __restrict__ proj) {
  __shared__ __align__(16) unsigned short Atile[2][128][32];
  __shared__ __align__(16) unsigned short Btile[2][64][32];
  const int wave = threadIdx.x >> 5;
  const int lane = threadIdx.x & 31;
  const int hf   = lane >> 4;
  const int n    = lane & 15;
  const int m0   = blockIdx.x * 128;
  const int n0   = blockIdx.y * 64;

  F8 acc[8];
#pragma unroll
  for (int j = 0; j < 8; ++j) acc[j].v = zero8();

  if (threadIdx.x < 32) {
    tdm_load_2d(lds_off(&Atile[0][0][0]), ao + (size_t)m0 * DM, 32, 128, DM);
    tdm_load_2d(lds_off(&Btile[0][0][0]), wot + (size_t)n0 * DM, 32, 64, DM);
  }
  int cur = 0;
  for (int k0 = 0; k0 < DM; k0 += 32) {
    if (threadIdx.x < 32) {
      if (k0 + 32 < DM) {
        tdm_load_2d(lds_off(&Atile[cur ^ 1][0][0]), ao + (size_t)m0 * DM + (k0 + 32), 32, 128, DM);
        tdm_load_2d(lds_off(&Btile[cur ^ 1][0][0]), wot + (size_t)n0 * DM + (k0 + 32), 32, 64, DM);
        __builtin_amdgcn_s_wait_tensorcnt(2);
      } else {
        __builtin_amdgcn_s_wait_tensorcnt(0);
      }
    }
    __syncthreads();
    const bf16x16 a0 = load_frag_a(&Atile[cur][0][0], wave * 32, 0, 32);
    const bf16x16 a1 = load_frag_a(&Atile[cur][0][0], wave * 32 + 16, 0, 32);
#pragma unroll
    for (int nt = 0; nt < 4; ++nt) {
      const bf16x16 bfr = load_frag_b(&Btile[cur][0][0], nt * 16, 0, 32);
      acc[nt].v     = wmma_bf16(a0, bfr, acc[nt].v);
      acc[4 + nt].v = wmma_bf16(a1, bfr, acc[4 + nt].v);
    }
    __syncthreads();
    cur ^= 1;
  }
#pragma unroll
  for (int sub = 0; sub < 2; ++sub) {
#pragma unroll
    for (int nt = 0; nt < 4; ++nt) {
      const int ncol = n0 + nt * 16 + n;
      const float bias = bo[ncol];
#pragma unroll
      for (int r = 0; r < 8; ++r) {
        const int m = m0 + wave * 32 + sub * 16 + r + hf * 8;
        proj[(size_t)m * DM + ncol] = acc[sub * 4 + nt].f[r] + bias;
      }
    }
  }
}

// ---------------- residual + LayerNorm: one block (256 thr) per row ----------------
__global__ void ln_kernel(const float* __restrict__ x, const float* __restrict__ proj,
                          const float* __restrict__ g, const float* __restrict__ be,
                          float* __restrict__ out) {
  const int row = blockIdx.x;
  const size_t base = (size_t)row * DM;
  float y[4], s1 = 0.0f, s2 = 0.0f;
#pragma unroll
  for (int j = 0; j < 4; ++j) {
    const int i = threadIdx.x + j * 256;
    y[j] = x[base + i] + proj[base + i];
    s1 += y[j]; s2 += y[j] * y[j];
  }
#pragma unroll
  for (int off = 1; off < 32; off <<= 1) {
    s1 += __shfl_xor(s1, off, 32);
    s2 += __shfl_xor(s2, off, 32);
  }
  __shared__ float r1[8], r2[8];
  if ((threadIdx.x & 31) == 0) { r1[threadIdx.x >> 5] = s1; r2[threadIdx.x >> 5] = s2; }
  __syncthreads();
  float t1 = 0.0f, t2 = 0.0f;
#pragma unroll
  for (int w = 0; w < 8; ++w) { t1 += r1[w]; t2 += r2[w]; }
  const float mu  = t1 * (1.0f / DM);
  const float var = t2 * (1.0f / DM) - mu * mu;
  const float rs  = rsqrtf(var + 1e-5f);
#pragma unroll
  for (int j = 0; j < 4; ++j) {
    const int i = threadIdx.x + j * 256;
    out[base + i] = (y[j] - mu) * rs * g[i] + be[i];
  }
}

// ---------------- launcher ----------------
extern "C" void kernel_launch(void* const* d_in, const int* in_sizes, int n_in,
                              void* d_out, int out_size, void* d_ws, size_t ws_size,
                              hipStream_t stream) {
  (void)in_sizes; (void)n_in; (void)out_size; (void)ws_size;
  const float* x    = (const float*)d_in[0];
  const int*   mask = (const int*)  d_in[1];
  const float* wq   = (const float*)d_in[2];
  const float* bq   = (const float*)d_in[3];
  const float* wk   = (const float*)d_in[4];
  const float* bk   = (const float*)d_in[5];
  const float* wv   = (const float*)d_in[6];
  const float* bv   = (const float*)d_in[7];
  const float* wo   = (const float*)d_in[8];
  const float* bo   = (const float*)d_in[9];
  const float* ab   = (const float*)d_in[10];
  const float* lng  = (const float*)d_in[11];
  const float* lnb  = (const float*)d_in[12];
  float* out = (float*)d_out;

  char* ws = (char*)d_ws;
  size_t off = 0;
  unsigned short* xb    = (unsigned short*)(ws + off); off += (size_t)BS * DM * 2;      // 8 MB
  unsigned short* wqkvt = (unsigned short*)(ws + off); off += (size_t)3 * DM * DM * 2;  // 6 MB
  unsigned short* wot   = (unsigned short*)(ws + off); off += (size_t)DM * DM * 2;      // 2 MB
  unsigned short* qbuf  = (unsigned short*)(ws + off); off += (size_t)BS * DM * 2;      // 8 MB
  unsigned short* kbuf  = (unsigned short*)(ws + off); off += (size_t)BS * DM * 2;      // 8 MB
  unsigned short* vtbuf = (unsigned short*)(ws + off); off += (size_t)BS * DM * 2;      // 8 MB
  unsigned short* aobuf = (unsigned short*)(ws + off); off += (size_t)BS * DM * 2;      // 8 MB
  float*          proj  = (float*)(ws + off);          off += (size_t)BS * DM * 4;      // 16 MB

  cvt_x_kernel<<<(BS * DM) / 256, 256, 0, stream>>>(x, xb);
  cvt_w_kernel<<<(DM * DM) / 256, 256, 0, stream>>>(wq, wk, wv, wo, wqkvt, wot);
  gemm_qkv_kernel<<<dim3(BS / 128, (3 * DM) / 64), 128, 0, stream>>>(
      xb, wqkvt, bq, bk, bv, qbuf, kbuf, vtbuf);
  attn_kernel<<<dim3(S_LEN / 64, NHEAD, BATCH), 128, 0, stream>>>(
      qbuf, kbuf, vtbuf, mask, ab, aobuf);
  gemm_out_kernel<<<dim3(BS / 128, DM / 64), 128, 0, stream>>>(aobuf, wot, bo, proj);
  ln_kernel<<<BS, 256, 0, stream>>>(x, proj, lng, lnb, out);
}
